// MultiHeadAttention3D_584115552857
// MI455X (gfx1250) — matrix operand order repaired and checked by reading, not compiled, not run on hardware
//
#include <hip/hip_runtime.h>
#include <hip/hip_bf16.h>

typedef __bf16 bf16;
typedef __attribute__((ext_vector_type(16))) __bf16 v16bf;
typedef __attribute__((ext_vector_type(8)))  float  v8f;
typedef __attribute__((ext_vector_type(4)))  unsigned int u32x4;
typedef __attribute__((ext_vector_type(4)))  int v4i;

#define LOG2E 1.44269504088896340736f

// Problem sizes
#define C_DIM 512
#define N_DIM 4096
#define NH    8
#define HD    64

// Workspace layout (bytes)
#define OFF_WQ  0
#define OFF_WK  (C_DIM*C_DIM*2)
#define OFF_WV  (2*C_DIM*C_DIM*2)
#define OFF_XT  (3*C_DIM*C_DIM*2)
#define OFF_Q   (OFF_XT + N_DIM*C_DIM*2)
#define OFF_K   (OFF_Q + NH*N_DIM*HD*2)
#define OFF_V   (OFF_K + NH*N_DIM*HD*2)

#define AS1 __attribute__((address_space(1)))
#define AS3 __attribute__((address_space(3)))

#if __has_builtin(__builtin_amdgcn_global_load_async_to_lds_b128)
#define ASYNC_LDS 1
#else
#define ASYNC_LDS 0
#endif

#if ASYNC_LDS
#if __has_builtin(__builtin_amdgcn_s_wait_asynccnt)
#define WAIT_ASYNC(n) __builtin_amdgcn_s_wait_asynccnt(n)
#else
#define WAIT_ASYNC(n) asm volatile("s_wait_asynccnt %0" :: "i"(n) : "memory")
#endif
#else
#define WAIT_ASYNC(n) do {} while (0)
#endif

union FragU {
    v16bf v;
    struct { u32x4 lo, hi; } u;
};

static __device__ __forceinline__ v8f wmma_bf16(const FragU& a, const FragU& b, v8f c) {
    return __builtin_amdgcn_wmma_f32_16x16x32_bf16(
        false, a.v, false, b.v, (short)0, c, false, false);
}

static __device__ __forceinline__ float redmax16(float v) {
    v = fmaxf(v, __shfl_xor(v, 1, 32));
    v = fmaxf(v, __shfl_xor(v, 2, 32));
    v = fmaxf(v, __shfl_xor(v, 4, 32));
    v = fmaxf(v, __shfl_xor(v, 8, 32));
    return v;
}
static __device__ __forceinline__ float redsum16(float v) {
    v += __shfl_xor(v, 1, 32);
    v += __shfl_xor(v, 2, 32);
    v += __shfl_xor(v, 4, 32);
    v += __shfl_xor(v, 8, 32);
    return v;
}

// 16-byte global -> LDS copy: async (ASYNCcnt-tracked, no VGPR round trip) when
// the gfx1250 async-LDS builtin is available, else load+store fallback.
// Builtin signature (from hipcc diagnostic): (v4i AS1* src, v4i AS3* dst, Ii, Ii)
static __device__ __forceinline__ void stage16(const bf16* g, bf16* l) {
#if ASYNC_LDS
    __builtin_amdgcn_global_load_async_to_lds_b128((AS1 v4i*)g, (AS3 v4i*)l, 0, 0);
#else
    *(u32x4*)l = *(const u32x4*)g;
#endif
}

// ---------------------------------------------------------------------------
// Phase 0: convert weights to bf16 and build X^T (bf16, [s][c]) in workspace
// ---------------------------------------------------------------------------
__global__ void mha3d_cvt_kernel(const float* __restrict__ x,
                                 const float* __restrict__ Wq,
                                 const float* __restrict__ Wk,
                                 const float* __restrict__ Wv,
                                 bf16* __restrict__ wq_bf,
                                 bf16* __restrict__ wk_bf,
                                 bf16* __restrict__ wv_bf,
                                 bf16* __restrict__ xt_bf) {
    int tid = blockIdx.x * blockDim.x + threadIdx.x;
    int stride = gridDim.x * blockDim.x;
    for (int i = tid; i < C_DIM * C_DIM; i += stride) {
        wq_bf[i] = (bf16)Wq[i];
        wk_bf[i] = (bf16)Wk[i];
        wv_bf[i] = (bf16)Wv[i];
    }
    for (int i = tid; i < C_DIM * N_DIM; i += stride) {
        int c = i >> 12;        // i / 4096
        int s = i & (N_DIM - 1);
        xt_bf[s * C_DIM + c] = (bf16)x[i];  // coalesced read, L2-buffered write
    }
}

// ---------------------------------------------------------------------------
// Phase 1: Q/K/V projections. One block per 16-column s-tile; 8 waves each
// own 64 output rows for all three matrices (12 16x16 accum tiles / wave).
// Q,K stored [h][n][d]; V stored transposed [h][d][n].
// ---------------------------------------------------------------------------
__global__ __launch_bounds__(256) void mha3d_proj_kernel(
    const bf16* __restrict__ wq, const bf16* __restrict__ wk,
    const bf16* __restrict__ wv, const bf16* __restrict__ xt,
    const float* __restrict__ bq, const float* __restrict__ bk,
    const float* __restrict__ bv,
    bf16* __restrict__ q_ws, bf16* __restrict__ k_ws, bf16* __restrict__ v_ws)
{
    const int lane = threadIdx.x & 31;
    const int wave = threadIdx.x >> 5;
    const int lr   = lane & 15;
    const int hs   = lane >> 4;          // half-select within the wave
    const int s0   = blockIdx.x * 16;

    const bf16* wmat[3] = { wq, wk, wv };

    v8f acc[12];
    const v8f vzero = {0.f, 0.f, 0.f, 0.f, 0.f, 0.f, 0.f, 0.f};
#pragma unroll
    for (int t = 0; t < 12; ++t) acc[t] = vzero;

    for (int c0 = 0; c0 < C_DIM; c0 += 32) {
        // B fragment: X^T, lane<16 holds K=c0..c0+15 (32B contiguous)
        FragU bfr;
        const bf16* bp = xt + (s0 + lr) * C_DIM + c0 + hs * 16;
        bfr.u.lo = *(const u32x4*)(bp);
        bfr.u.hi = *(const u32x4*)(bp + 8);
#pragma unroll
        for (int m = 0; m < 3; ++m) {
            // batch the 4 A-fragment loads into distinct registers so the
            // compiler can clause them and use partial s_wait_loadcnt
            FragU afr[4];
#pragma unroll
            for (int ot = 0; ot < 4; ++ot) {
                const int o0 = wave * 64 + ot * 16;
                const bf16* ap = wmat[m] + (o0 + lr) * C_DIM + c0 + hs * 16;
                afr[ot].u.lo = *(const u32x4*)(ap);
                afr[ot].u.hi = *(const u32x4*)(ap + 8);
            }
#pragma unroll
            for (int ot = 0; ot < 4; ++ot)
                acc[m * 4 + ot] = wmma_bf16(afr[ot], bfr, acc[m * 4 + ot]);
        }
    }

    const float* bias[3] = { bq, bk, bv };
#pragma unroll
    for (int m = 0; m < 3; ++m) {
#pragma unroll
        for (int ot = 0; ot < 4; ++ot) {
            const int o0 = wave * 64 + ot * 16;
#pragma unroll
            for (int r = 0; r < 8; ++r) {
                const int o = o0 + r + 8 * hs;       // output channel (row)
                const int n = s0 + lr;               // spatial position (col)
                const float val = acc[m * 4 + ot][r] + bias[m][o];
                const int h = o >> 6, d = o & 63;
                if (m == 0)      q_ws[h * (N_DIM * HD) + n * HD + d] = (bf16)val;
                else if (m == 1) k_ws[h * (N_DIM * HD) + n * HD + d] = (bf16)val;
                else             v_ws[h * (HD * N_DIM) + d * N_DIM + n] = (bf16)val;
            }
        }
    }
}

// ---------------------------------------------------------------------------
// Phase 2: fused flash attention. Grid = 8 heads x 32 q-blocks of 128 rows.
// Block = 8 waves, each wave owns 16 q rows; K/V tiles (Bc=32) double-buffered
// in LDS via async global->LDS copies (ASYNCcnt) when available.
// ---------------------------------------------------------------------------
__global__ __launch_bounds__(256) void mha3d_attn_kernel(
    const bf16* __restrict__ q_ws, const bf16* __restrict__ k_ws,
    const bf16* __restrict__ v_ws, const float* __restrict__ x,
    const float* __restrict__ gamma_p, float* __restrict__ out)
{
    __shared__ bf16 K_lds[2][32 * 64];   // [buf][kv_local][d]
    __shared__ bf16 Vt_lds[2][64 * 32];  // [buf][d][kv_local]
    __shared__ bf16 P_lds[8][16][32];    // per-wave P relayout buffer

    const int t    = threadIdx.x;
    const int lane = t & 31;
    const int wave = t >> 5;
    const int lr   = lane & 15;
    const int hs   = lane >> 4;
    const int h    = blockIdx.x >> 5;
    const int q0   = (blockIdx.x & 31) * 128 + wave * 16;

    const bf16* qh = q_ws + h * (N_DIM * HD);
    const bf16* kh = k_ws + h * (N_DIM * HD);
    const bf16* vh = v_ws + h * (HD * N_DIM);

    // per-thread staging coordinates (16B each into K and Vt tiles)
    const int kr = t >> 3, kc = (t & 7) * 8;
    const int vr = t >> 2, vc = (t & 3) * 8;

    // Q fragments (d 0..31 and 32..63), resident in VGPRs for whole kernel
    FragU qf0, qf1;
    {
        const bf16* qp = qh + (q0 + lr) * HD + hs * 16;
        qf0.u.lo = *(const u32x4*)(qp);
        qf0.u.hi = *(const u32x4*)(qp + 8);
        qf1.u.lo = *(const u32x4*)(qp + 32);
        qf1.u.hi = *(const u32x4*)(qp + 40);
    }

    const v8f vzero = {0.f, 0.f, 0.f, 0.f, 0.f, 0.f, 0.f, 0.f};
    v8f Oacc[4];
#pragma unroll
    for (int f = 0; f < 4; ++f) Oacc[f] = vzero;
    float m_run[8], l_run[8];
#pragma unroll
    for (int r = 0; r < 8; ++r) { m_run[r] = -3.0e38f; l_run[r] = 0.f; }

    const int NT = N_DIM / 32;

    // prologue: prefetch tile 0 into buffer 0
    stage16(kh + kr * HD + kc,  &K_lds[0][kr * HD + kc]);
    stage16(vh + vr * N_DIM + vc, &Vt_lds[0][vr * 32 + vc]);

    for (int it = 0; it < NT; ++it) {
        const int cur = it & 1;
        const int j   = it * 32;

        if (it + 1 < NT) {
            // prefetch next tile into the other buffer, keep it in flight
            stage16(kh + (j + 32 + kr) * HD + kc,      &K_lds[cur ^ 1][kr * HD + kc]);
            stage16(vh + vr * N_DIM + (j + 32) + vc,   &Vt_lds[cur ^ 1][vr * 32 + vc]);
            WAIT_ASYNC(2);   // current tile's 2 async ops complete (in-order)
        } else {
            WAIT_ASYNC(0);
        }
        __syncthreads();

        const bf16* Kc  = &K_lds[cur][0];
        const bf16* Vtc = &Vt_lds[cur][0];

        // S tiles: S_a = Q.K^T over kv rows 0..15, S_b over 16..31
        v8f Sa = vzero, Sb = vzero;
        {
            FragU b0, b1;
            b0.u.lo = *(const u32x4*)(Kc + lr * HD + hs * 16);
            b0.u.hi = *(const u32x4*)(Kc + lr * HD + hs * 16 + 8);
            b1.u.lo = *(const u32x4*)(Kc + lr * HD + 32 + hs * 16);
            b1.u.hi = *(const u32x4*)(Kc + lr * HD + 32 + hs * 16 + 8);
            Sa = wmma_bf16(qf0, b0, Sa);
            Sa = wmma_bf16(qf1, b1, Sa);
            b0.u.lo = *(const u32x4*)(Kc + (16 + lr) * HD + hs * 16);
            b0.u.hi = *(const u32x4*)(Kc + (16 + lr) * HD + hs * 16 + 8);
            b1.u.lo = *(const u32x4*)(Kc + (16 + lr) * HD + 32 + hs * 16);
            b1.u.hi = *(const u32x4*)(Kc + (16 + lr) * HD + 32 + hs * 16 + 8);
            Sb = wmma_bf16(qf0, b0, Sb);
            Sb = wmma_bf16(qf1, b1, Sb);
        }

        // online softmax: rows r+8*hs live across the 16-lane half-group
        float alpha[8];
#pragma unroll
        for (int r = 0; r < 8; ++r) {
            const float rmax = redmax16(fmaxf(Sa[r], Sb[r]));
            const float mn   = fmaxf(m_run[r], rmax);
            alpha[r] = exp2f((m_run[r] - mn) * LOG2E);
            m_run[r] = mn;
            const float pa = exp2f((Sa[r] - mn) * LOG2E);
            const float pb = exp2f((Sb[r] - mn) * LOG2E);
            l_run[r] = l_run[r] * alpha[r] + redsum16(pa + pb);
            const int ro = r + 8 * hs;
            P_lds[wave][ro][lr]      = (bf16)pa;
            P_lds[wave][ro][16 + lr] = (bf16)pb;
        }
#pragma unroll
        for (int f = 0; f < 4; ++f)
#pragma unroll
            for (int r = 0; r < 8; ++r) Oacc[f][r] *= alpha[r];

        // P as A-fragment (same-wave LDS: compiler orders via DScnt)
        FragU pf;
        {
            const bf16* pp = &P_lds[wave][lr][hs * 16];
            pf.u.lo = *(const u32x4*)(pp);
            pf.u.hi = *(const u32x4*)(pp + 8);
        }
        // O += P.V : 4 d-column fragments of V^T
#pragma unroll
        for (int f = 0; f < 4; ++f) {
            FragU vf;
            const bf16* vp = Vtc + (f * 16 + lr) * 32 + hs * 16;
            vf.u.lo = *(const u32x4*)(vp);
            vf.u.hi = *(const u32x4*)(vp + 8);
            Oacc[f] = wmma_bf16(pf, vf, Oacc[f]);
        }
        __syncthreads();
    }

    // epilogue: y[n*512 + h*64 + d] = gamma * O/l + x[same flat index]
    const float g = gamma_p[0];
#pragma unroll
    for (int f = 0; f < 4; ++f) {
#pragma unroll
        for (int r = 0; r < 8; ++r) {
            const int n = q0 + r + 8 * hs;
            const int d = f * 16 + lr;
            const int i = n * C_DIM + h * HD + d;
            out[i] = g * (Oacc[f][r] / l_run[r]) + x[i];
        }
    }
}

extern "C" void kernel_launch(void* const* d_in, const int* in_sizes, int n_in,
                              void* d_out, int out_size, void* d_ws, size_t ws_size,
                              hipStream_t stream) {
    const float* x     = (const float*)d_in[0];
    const float* Wq    = (const float*)d_in[1];
    const float* bq    = (const float*)d_in[2];
    const float* Wk    = (const float*)d_in[3];
    const float* bk    = (const float*)d_in[4];
    const float* Wv    = (const float*)d_in[5];
    const float* bv    = (const float*)d_in[6];
    const float* gamma = (const float*)d_in[7];
    float* out = (float*)d_out;

    char* ws = (char*)d_ws;
    bf16* wq_bf = (bf16*)(ws + OFF_WQ);
    bf16* wk_bf = (bf16*)(ws + OFF_WK);
    bf16* wv_bf = (bf16*)(ws + OFF_WV);
    bf16* xt_bf = (bf16*)(ws + OFF_XT);
    bf16* q_ws  = (bf16*)(ws + OFF_Q);
    bf16* k_ws  = (bf16*)(ws + OFF_K);
    bf16* v_ws  = (bf16*)(ws + OFF_V);

    mha3d_cvt_kernel<<<1024, 256, 0, stream>>>(x, Wq, Wk, Wv,
                                               wq_bf, wk_bf, wv_bf, xt_bf);
    mha3d_proj_kernel<<<N_DIM / 16, 256, 0, stream>>>(wq_bf, wk_bf, wv_bf, xt_bf,
                                                      bq, bk, bv, q_ws, k_ws, v_ws);
    mha3d_attn_kernel<<<NH * (N_DIM / 128), 256, 0, stream>>>(q_ws, k_ws, v_ws,
                                                              x, gamma, out);
}
